// AffineCouplingBlock_6287832121753
// MI455X (gfx1250) — matrix operand order repaired and checked by reading, not compiled, not run on hardware
//
#include <hip/hip_runtime.h>
#include <hip/hip_bf16.h>
#include <hip/hip_fp16.h>

typedef __attribute__((ext_vector_type(16))) _Float16 v16h;
typedef __attribute__((ext_vector_type(8)))  float    v8f;
typedef __attribute__((ext_vector_type(4)))  float    v4f;

#define GAS __attribute__((address_space(1)))
#define LAS __attribute__((address_space(3)))
typedef const GAS _Float16* gch16;
typedef const GAS float*    gcf32;
typedef GAS float*          gf32;
typedef const GAS v4f*      gcv4f;
typedef LAS _Float16*       lh16;
typedef const LAS _Float16* lch16;
typedef const LAS v4f*      lcv4f;

#define B_SZ   64
#define T_PTS  4096
#define CCH    64
#define CONDN  16
#define HID    128
#define INDIM  81
#define K0PAD  96
#define TP     128          // points per block (4 waves x 32 points each)
#define NTHREADS 128

// ---- workspace layout (bytes) ----
static constexpr size_t WS_WH0 = 0;                                             // 4*128*96  f16
static constexpr size_t WS_WH1 = WS_WH0 + (size_t)4*HID*K0PAD*sizeof(_Float16); // 4*3*128*128 f16
static constexpr size_t WS_WH2 = WS_WH1 + (size_t)4*3*HID*HID*sizeof(_Float16); // 4*128 f16
static constexpr size_t WS_ZA  = WS_WH2 + (size_t)4*HID*sizeof(_Float16);       // B*T f32

// -------------------- prep: f32 weights -> padded f16, zero log_det --------------------
__global__ void prep_kernel(const float* __restrict__ W0, const float* __restrict__ W1,
                            const float* __restrict__ W2,
                            _Float16* __restrict__ Wh0, _Float16* __restrict__ Wh1,
                            _Float16* __restrict__ Wh2, float* __restrict__ logdet) {
  int i = blockIdx.x * blockDim.x + threadIdx.x;
  gcf32 W0g = (gcf32)W0; gcf32 W1g = (gcf32)W1; gcf32 W2g = (gcf32)W2;
  GAS _Float16* Wh0g = (GAS _Float16*)Wh0;
  GAS _Float16* Wh1g = (GAS _Float16*)Wh1;
  GAS _Float16* Wh2g = (GAS _Float16*)Wh2;
  if (i < 4 * HID * K0PAD) {                       // [4][128][96], pad K 81..95 with 0
    int k = i % K0PAD;
    int n = (i / K0PAD) % HID;
    int m = i / (K0PAD * HID);
    Wh0g[i] = (k < INDIM) ? (_Float16)W0g[((size_t)m * HID + n) * INDIM + k] : (_Float16)0.0f;
  }
  if (i < 4 * 3 * HID * HID) Wh1g[i] = (_Float16)W1g[i];
  if (i < 4 * HID)           Wh2g[i] = (_Float16)W2g[i];
  if (i < B_SZ)              ((gf32)logdet)[i] = 0.0f;
}

// -------------------- WMMA fragment helpers --------------------
// A fragment: 16x32 f16, rows = points (LDS). lane m(0..15)/m+16 hold row m,
// kh = lane>>4 selects K {kh*8..+7} in VGPR0-3 and {16+kh*8..+7} in VGPR4-7.
__device__ inline v16h load_a_frag(lch16 base, int stride, int kb, int lane) {
  int m = lane & 15, kh = (lane >> 4) & 1;
  lch16 p0 = base + (size_t)m * stride + kb + kh * 8;
  v16h r;
  ((v4f*)&r)[0] = ((lcv4f)p0)[0];           // halves 0..7  (VGPR0-3)
  ((v4f*)&r)[1] = ((lcv4f)(p0 + 16))[0];    // halves 8..15 (VGPR4-7)
  return r;
}

// B fragment: 32x16 f16 (KxN), weights in GLOBAL memory, stored [N][Kstride] row-major.
// lane col n, kh = lane>>4: K {kb+kh*8..+7} in halves 0..7 and {kb+16+kh*8..+7} in halves 8..15.
__device__ inline v16h load_b_frag(gch16 wrows, int kstride, int kb, int lane) {
  int n = lane & 15, kh = (lane >> 4) & 1;
  gch16 p = wrows + (size_t)n * kstride + kb + kh * 8;
  v16h r;
  ((v4f*)&r)[0] = ((gcv4f)p)[0];
  ((v4f*)&r)[1] = ((gcv4f)(p + 16))[0];
  return r;
}

__device__ inline v8f wmma_f16(v16h a, v16h b, v8f c) {
  return __builtin_amdgcn_wmma_f32_16x16x32_f16(false, a, false, b, (short)0, c, false, false);
}

// One dense layer for this wave's 32 points (two 16-row blocks):
//   Hout[32 x 128] = silu(Ain[32 x kin] @ W^T + b)
// N is processed in two sequential halves of 64 columns so the live accumulator
// set is 2x4 v8f = 64 VGPRs (stays under the 256-VGPR direct window -> no
// s_set_vgpr_msb churn). Each B fragment is reused by TWO WMMAs (row blocks).
__device__ inline void mlp_layer(lch16 Ain, int kin,
                                 gch16 Wg, gcf32 bg,
                                 lh16 Hout, int wave, int lane) {
  lch16 arow0 = Ain + (size_t)(wave * 32) * kin;        // rows 32w..32w+15
  lch16 arow1 = arow0 + (size_t)16 * kin;               // rows 32w+16..32w+31
  int col = lane & 15, kh = (lane >> 4) & 1;

#pragma unroll 1
  for (int half = 0; half < 2; ++half) {
    gch16 Wh = Wg + (size_t)(half * 64) * kin;          // 4 n-tiles of this half
    v8f acc0[4], acc1[4];
#pragma unroll
    for (int n = 0; n < 4; ++n)
#pragma unroll
      for (int r = 0; r < 8; ++r) { acc0[n][r] = 0.0f; acc1[n][r] = 0.0f; }

    v16h bf = load_b_frag(Wh, kin, 0, lane);            // prefetch (k=0, n=0)
    for (int kb = 0; kb < kin; kb += 32) {
      v16h a0 = load_a_frag(arow0, kin, kb, lane);
      v16h a1 = load_a_frag(arow1, kin, kb, lane);
#pragma unroll
      for (int n = 0; n < 4; ++n) {
        v16h bcur = bf;
        // issue the next B-fragment load before consuming the current one
        int nn  = (n + 1) & 3;
        int kbn = (n == 3) ? ((kb + 32 < kin) ? kb + 32 : kb) : kb;
        bf = load_b_frag(Wh + (size_t)(nn * 16) * kin, kin, kbn, lane);
        acc0[n] = wmma_f16(a0, bcur, acc0[n]);
        acc1[n] = wmma_f16(a1, bcur, acc1[n]);
      }
    }
    // C/D layout: VGPR r -> row = r + (lane>>4)*8, col = lane&15
#pragma unroll
    for (int n = 0; n < 4; ++n) {
      float bias = bg[half * 64 + n * 16 + col];
#pragma unroll
      for (int r = 0; r < 8; ++r) {
        float v0 = acc0[n][r] + bias;
        float v1 = acc1[n][r] + bias;
        v0 = v0 / (1.0f + __expf(-v0));       // SiLU
        v1 = v1 / (1.0f + __expf(-v1));
        int rowbase = wave * 32 + kh * 8 + r;
        int colo = half * 64 + n * 16 + col;
        Hout[(size_t)(rowbase)      * HID + colo] = (_Float16)v0;
        Hout[(size_t)(rowbase + 16) * HID + colo] = (_Float16)v1;
      }
    }
  }
}

// Full 4-layer MLP + scalar head for this wave's 32 points.
// Returns raw head output; lane L holds the value for point wave*32 + L.
__device__ inline float run_mlp(lch16 comb,
                                gch16 Wh0, gch16 Wh1, gch16 Wh2,
                                gcf32 b0, gcf32 b1, float b2v,
                                lh16 hb0, lh16 hb1, int idx, int wave, int lane) {
  mlp_layer(comb, K0PAD, Wh0 + (size_t)idx * HID * K0PAD, b0 + idx * HID, hb0, wave, lane);
  mlp_layer(hb0, HID, Wh1 + (size_t)(idx * 3 + 0) * HID * HID, b1 + (idx * 3 + 0) * HID, hb1, wave, lane);
  mlp_layer(hb1, HID, Wh1 + (size_t)(idx * 3 + 1) * HID * HID, b1 + (idx * 3 + 1) * HID, hb0, wave, lane);
  mlp_layer(hb0, HID, Wh1 + (size_t)(idx * 3 + 2) * HID * HID, b1 + (idx * 3 + 2) * HID, hb1, wave, lane);
  // head: out = h . W2 + b2 ; one lane per point, full 128-wide dot
  int prow = wave * 32 + lane;
  lch16 hrow = hb1 + (size_t)prow * HID;
  gch16 wrow = Wh2 + (size_t)idx * HID;
  float sum = 0.0f;
#pragma unroll
  for (int cch = 0; cch < 16; ++cch) {
    v4f hv = ((lcv4f)hrow)[cch];
    v4f wv = ((gcv4f)wrow)[cch];
    const _Float16* he = (const _Float16*)&hv;
    const _Float16* we = (const _Float16*)&wv;
#pragma unroll
    for (int e = 0; e < 8; ++e) sum += (float)he[e] * (float)we[e];
  }
  return sum + b2v;
}

// -------------------- one coupling pass --------------------
__global__ __launch_bounds__(NTHREADS) void coupling_pass(
    const float* __restrict__ xp, const float* __restrict__ wp, const float* __restrict__ condp,
    const _Float16* __restrict__ Wh0p, const _Float16* __restrict__ Wh1p, const _Float16* __restrict__ Wh2p,
    const float* __restrict__ b0p, const float* __restrict__ b1p, const float* __restrict__ b2p,
    float* __restrict__ zap, float* __restrict__ outp, float* __restrict__ logdet, int passB) {
  extern __shared__ _Float16 smem[];
  lh16 combL = (lh16)smem;                 // [TP][K0PAD]
  lh16 hb0   = combL + TP * K0PAD;         // [TP][HID]
  lh16 hb1   = hb0 + TP * HID;             // [TP][HID]

  // explicit global address-space views (force global_load/store, not flat)
  gcf32 x    = (gcf32)xp;
  gcf32 w    = (gcf32)wp;
  gcf32 cond = (gcf32)condp;
  gch16 Wh0  = (gch16)Wh0p;
  gch16 Wh1  = (gch16)Wh1p;
  gch16 Wh2  = (gch16)Wh2p;
  gcf32 b0   = (gcf32)b0p;
  gcf32 b1   = (gcf32)b1p;
  gcf32 b2   = (gcf32)b2p;
  gf32  za   = (gf32)zap;
  gf32  out  = (gf32)outp;

  const int tid   = threadIdx.x;          // 0..127 : one thread per point
  const int lane  = tid & 31;
  const int wave  = tid >> 5;
  const int pBase = blockIdx.x * TP;
  const int b     = pBase >> 12;          // T_PTS = 4096 points per batch
  const int p     = pBase + tid;

  // ---- gather + comb build: 1 thread per point (same wave as its MLP rows) ----
  float gx, gy, zcur, zkeep;
  if (passB) {
    float a0 = za[p];
    gx = a0; gy = x[2 * p + 1];
    zkeep = a0; zcur = x[2 * p + 1];
  } else {
    gx = x[2 * p]; gy = x[2 * p + 1];
    zkeep = gy; zcur = gx;
  }
  // grid = cur*2-1 ; ix = (grid+1)*0.5*(W-1) = cur*63
  float ix = gx * 63.0f, iy = gy * 63.0f;
  float ix0f = floorf(ix), iy0f = floorf(iy);
  float fx = ix - ix0f, fy = iy - iy0f;
  int ix0 = (int)ix0f, iy0 = (int)iy0f;
  int ix1 = ix0 + 1, iy1 = iy0 + 1;
  bool vx0 = (ix0 >= 0) && (ix0 <= 63), vx1 = (ix1 >= 0) && (ix1 <= 63);
  bool vy0 = (iy0 >= 0) && (iy0 <= 63), vy1 = (iy1 >= 0) && (iy1 <= 63);
  int xc0 = min(max(ix0, 0), 63), xc1 = min(max(ix1, 0), 63);
  int yc0 = min(max(iy0, 0), 63), yc1 = min(max(iy1, 0), 63);
  float w00 = (vy0 && vx0) ? (1.0f - fy) * (1.0f - fx) : 0.0f;
  float w01 = (vy0 && vx1) ? (1.0f - fy) * fx          : 0.0f;
  float w10 = (vy1 && vx0) ? fy * (1.0f - fx)          : 0.0f;
  float w11 = (vy1 && vx1) ? fy * fx                   : 0.0f;

  gcf32 plane = w + (size_t)b * CCH * 4096;
  lh16 crow = combL + (size_t)tid * K0PAD;
#pragma unroll 8
  for (int c = 0; c < CCH; ++c) {
    gcf32 pp = plane + (size_t)c * 4096;
    float v = w00 * pp[yc0 * 64 + xc0] + w01 * pp[yc0 * 64 + xc1] +
              w10 * pp[yc1 * 64 + xc0] + w11 * pp[yc1 * 64 + xc1];
    crow[1 + c] = (_Float16)v;
  }
  crow[0] = (_Float16)zkeep;
#pragma unroll
  for (int j = 0; j < CONDN; ++j) crow[1 + CCH + j] = (_Float16)cond[b * CONDN + j];
#pragma unroll
  for (int j = INDIM; j < K0PAD; ++j) crow[j] = (_Float16)0.0f;   // zero pad 81..95
  // No __syncthreads needed: each wave only reads/writes its own 32 LDS rows,
  // and per-wave DS ordering (DScnt) covers the store->load dependency.

  const int idx_s = passB ? 2 : 0;
  const int idx_t = passB ? 3 : 1;
  float sraw = run_mlp(combL, Wh0, Wh1, Wh2, b0, b1, b2[idx_s], hb0, hb1, idx_s, wave, lane);
  float traw = run_mlp(combL, Wh0, Wh1, Wh2, b0, b1, b2[idx_t], hb0, hb1, idx_t, wave, lane);

  float sc = fminf(fmaxf(sraw, -15.0f), 15.0f);
  float e2 = __expf(2.0f * sc);
  float s  = ((e2 - 1.0f) / (e2 + 1.0f)) * 1.5f;   // tanh * S_MAX

  float zn = zcur * __expf(s) + traw;
  if (passB) {
    out[2 * p + 1] = zn;
  } else {
    out[2 * p] = zn;
    za[p] = zn;
  }
  // log_det: wave reduction of s over this wave's 32 points, one atomic per wave
  float contrib = s;
#pragma unroll
  for (int off = 1; off < 32; off <<= 1) contrib += __shfl_xor(contrib, off);
  if (lane == 0) atomicAdd(&logdet[b], contrib);
}

// -------------------- launch --------------------
extern "C" void kernel_launch(void* const* d_in, const int* in_sizes, int n_in,
                              void* d_out, int out_size, void* d_ws, size_t ws_size,
                              hipStream_t stream) {
  (void)in_sizes; (void)n_in; (void)out_size; (void)ws_size;
  const float* x    = (const float*)d_in[0];
  const float* w    = (const float*)d_in[1];
  const float* cond = (const float*)d_in[2];
  const float* W0   = (const float*)d_in[3];
  const float* b0   = (const float*)d_in[4];
  const float* W1   = (const float*)d_in[5];
  const float* b1   = (const float*)d_in[6];
  const float* W2   = (const float*)d_in[7];
  const float* b2   = (const float*)d_in[8];

  char* ws = (char*)d_ws;
  _Float16* Wh0 = (_Float16*)(ws + WS_WH0);
  _Float16* Wh1 = (_Float16*)(ws + WS_WH1);
  _Float16* Wh2 = (_Float16*)(ws + WS_WH2);
  float*    za  = (float*)(ws + WS_ZA);

  float* out    = (float*)d_out;
  float* logdet = out + (size_t)B_SZ * T_PTS * 2;

  prep_kernel<<<768, 256, 0, stream>>>(W0, W1, W2, Wh0, Wh1, Wh2, logdet);

  dim3 grid(B_SZ * T_PTS / TP);                                  // 2048 blocks
  size_t lds = (size_t)(TP * K0PAD + 2 * TP * HID) * sizeof(_Float16);  // 90112 B
  coupling_pass<<<grid, NTHREADS, lds, stream>>>(x, w, cond, Wh0, Wh1, Wh2, b0, b1, b2,
                                                 za, out, logdet, 0);
  coupling_pass<<<grid, NTHREADS, lds, stream>>>(x, w, cond, Wh0, Wh1, Wh2, b0, b1, b2,
                                                 za, out, logdet, 1);
}
